// MultiHeadAttention_19920058318981
// MI455X (gfx1250) — matrix operand order repaired and checked by reading, not compiled, not run on hardware
//
#include <hip/hip_runtime.h>

// MultiHeadAttention for MI455X (gfx1250): f16 WMMA (16x16x32) GEMMs with fp32
// accumulate, online-softmax causal attention, full (B,H,S,S) weights emission.
// Weights/out use non-temporal stores so the 268MB weights stream does not
// thrash the 192MB L2 where Q/K/V (25MB f16) are kept resident.

#define B_  4
#define S_  1024
#define D_  1024
#define H_  16
#define DK_ 64

typedef _Float16 half8_t  __attribute__((ext_vector_type(8)));
typedef _Float16 half16_t __attribute__((ext_vector_type(16)));
typedef float    float8_t __attribute__((ext_vector_type(8)));

#define NEG_BIG (-1.0e30f)

__device__ __forceinline__ float8_t wmma_f16(half16_t a, half16_t b, float8_t c) {
  // D = A(16x32) * B(32x16) + C, fp32 accumulate
  return __builtin_amdgcn_wmma_f32_16x16x32_f16(false, a, false, b, (short)0, c,
                                                false, false);
}

// A fragment: 16(M) x 32(K). Source row-major, leading dim ld (halves).
// Per ISA: lane m (0..15) / m+16 holds two contiguous 8-half chunks:
//   elems 0..7  -> K = k0 + 8*hs + 0..7
//   elems 8..15 -> K = k0 + 16 + 8*hs + 0..7
__device__ __forceinline__ half16_t load_a_frag(const _Float16* base, int ld,
                                                int row0, int k0, int lane) {
  const int m  = lane & 15;
  const int hs = lane >> 4;
  const _Float16* p = base + (size_t)(row0 + m) * ld + k0 + hs * 8;
  half8_t lo = *(const half8_t*)(p);
  half8_t hi = *(const half8_t*)(p + 16);
  half16_t a;
#pragma unroll
  for (int i = 0; i < 8; ++i) { a[i] = lo[i]; a[i + 8] = hi[i]; }
  return a;
}

// B fragment: 32(K) x 16(N). Source stored N-major (row n holds K contiguous):
// lane n (0..15) / n+16 holds two 8-half chunks:
//   elems 0..7  -> K = k0 + 8*hs + 0..7
//   elems 8..15 -> K = k0 + 16 + 8*hs + 0..7
__device__ __forceinline__ half16_t load_b_frag(const _Float16* base, int ld,
                                                int col0, int k0, int lane) {
  const int n  = lane & 15;
  const int hs = lane >> 4;
  const _Float16* p = base + (size_t)(col0 + n) * ld + k0 + hs * 8;
  half8_t lo = *(const half8_t*)(p);
  half8_t hi = *(const half8_t*)(p + 16);
  half16_t b;
#pragma unroll
  for (int i = 0; i < 8; ++i) { b[i] = lo[i]; b[i + 8] = hi[i]; }
  return b;
}

// ---------------------------------------------------------------------------
// Kernel 1: f32 -> f16 conversion (optionally scaled). Read-once sources are
// loaded non-temporally.
// ---------------------------------------------------------------------------
__global__ void cvt_f32_f16(const float* __restrict__ src,
                            _Float16* __restrict__ dst, int n, float scale) {
  int i = blockIdx.x * blockDim.x + threadIdx.x;
  if (i < n) {
    float v = __builtin_nontemporal_load(src + i);
    dst[i] = (_Float16)(v * scale);
  }
}

// ---------------------------------------------------------------------------
// Kernel 2: C(M=4096,N=1024) = A(M,K=1024) @ W(N,K)^T  via WMMA f16.
// 256 threads = 8 waves (4 M x 2 N); workgroup tile 256x128; wave tile 64x64
// (16 WMMAs / K-step -> ~42 FLOP per L2 byte, halving L2 traffic vs 64x32).
// mode 0: f16 dst[((b*H+h)*S + s)*DK + dk]      (Q,K per-head layout)
// mode 1: f16 dst[((b*H+h)*DK + dk)*S + s]      (V transposed per-head layout)
// mode 2: f32 dst[m*D + n]                      (final output projection, NT)
// ---------------------------------------------------------------------------
__global__ __launch_bounds__(256)
void gemm_x_wT(const _Float16* __restrict__ A, const _Float16* __restrict__ W,
               void* __restrict__ dst, int mode) {
  const int lane = threadIdx.x & 31;
  const int w    = threadIdx.x >> 5;
  const int row0 = blockIdx.y * 256 + (w >> 1) * 64;  // 4 wave-rows of 64
  const int col0 = blockIdx.x * 128 + (w & 1) * 64;   // 2 wave-cols of 64

  float8_t acc[4][4] = {};
  for (int kk = 0; kk < D_; kk += 32) {
    if (kk + 32 < D_) {
      // prefetch the next K-step one iteration ahead (global_prefetch_b8)
      __builtin_prefetch(&A[(size_t)(row0 + (lane & 15)) * D_ + kk + 32], 0, 1);
      __builtin_prefetch(&W[(size_t)(col0 + (lane & 15)) * D_ + kk + 32], 0, 1);
    }
    half16_t af[4], bf[4];
#pragma unroll
    for (int i = 0; i < 4; ++i) af[i] = load_a_frag(A, D_, row0 + 16 * i, kk, lane);
#pragma unroll
    for (int j = 0; j < 4; ++j) bf[j] = load_b_frag(W, D_, col0 + 16 * j, kk, lane);
#pragma unroll
    for (int i = 0; i < 4; ++i)
#pragma unroll
      for (int j = 0; j < 4; ++j) acc[i][j] = wmma_f16(af[i], bf[j], acc[i][j]);
  }

  const int laneHi = lane >> 4;
  const int cn     = lane & 15;
#pragma unroll
  for (int i = 0; i < 4; ++i) {
#pragma unroll
    for (int j = 0; j < 4; ++j) {
      const int mt = row0 + 16 * i;
      const int n  = col0 + 16 * j + cn;
#pragma unroll
      for (int e = 0; e < 8; ++e) {
        const int m = mt + e + 8 * laneHi;
        const float v = acc[i][j][e];
        if (mode == 2) {
          __builtin_nontemporal_store(v, &((float*)dst)[(size_t)m * D_ + n]);
        } else {
          const int b = m >> 10, s = m & (S_ - 1);
          const int h = n >> 6,  dk = n & (DK_ - 1);
          _Float16* o = (_Float16*)dst;
          if (mode == 0)
            o[(((size_t)(b * H_ + h)) * S_ + s) * DK_ + dk] = (_Float16)v;
          else
            o[(((size_t)(b * H_ + h)) * DK_ + dk) * S_ + s] = (_Float16)v;
        }
      }
    }
  }
}

// ---------------------------------------------------------------------------
// Kernel 3: causal masked attention per (b,h). 256 threads = 8 waves, each
// wave owns 16 query rows. Pass 1: fused online max+sum (flash-style).
// Pass 2: emit normalized weights (NT stores) + P@V via LDS re-layout.
// ---------------------------------------------------------------------------
__global__ __launch_bounds__(256)
void attn_kernel(const _Float16* __restrict__ qh,   // (B,H,S,DK) f16, scale folded
                 const _Float16* __restrict__ kh,   // (B,H,S,DK) f16
                 const _Float16* __restrict__ vt,   // (B,H,DK,S) f16 (transposed)
                 const unsigned char* __restrict__ q_pad,   // (B,S)
                 const unsigned char* __restrict__ kv_pad,  // (B,S)
                 float* __restrict__ w_out,         // (B,H,S,S) f32
                 _Float16* __restrict__ attn_h) {   // (B,S,D) f16
  __shared__ _Float16 plds[8][16 * 32];             // per-wave P re-layout slice

  const int lane   = threadIdx.x & 31;
  const int w      = threadIdx.x >> 5;
  const int bh     = blockIdx.y;
  const int b      = bh >> 4;
  const int h      = bh & 15;
  const int m0     = blockIdx.x * 128 + w * 16;     // this wave's 16 query rows
  const int laneHi = lane >> 4;
  const int cn     = lane & 15;

  const _Float16* qb = qh + (size_t)bh * S_ * DK_;
  const _Float16* kb = kh + (size_t)bh * S_ * DK_;
  const _Float16* vb = vt + (size_t)bh * DK_ * S_;

  const half16_t aq0 = load_a_frag(qb, DK_, m0, 0, lane);
  const half16_t aq1 = load_a_frag(qb, DK_, m0, 32, lane);

  const int tmax = m0 >> 4;  // last column tile touching the causal region

  // ---- pass 1: fused online per-row max and sum (per-lane partials) ----
  float rmax[8], rsum[8];
#pragma unroll
  for (int e = 0; e < 8; ++e) { rmax[e] = NEG_BIG; rsum[e] = 0.0f; }
  for (int t = 0; t <= tmax; ++t) {
    const int n0 = t << 4;
    half16_t bk0 = load_b_frag(kb, DK_, n0, 0, lane);
    half16_t bk1 = load_b_frag(kb, DK_, n0, 32, lane);
    float8_t c = {};
    c = wmma_f16(aq0, bk0, c);
    c = wmma_f16(aq1, bk1, c);
    const int  j   = n0 + cn;
    const bool kvp = kv_pad[b * S_ + j] != 0;
#pragma unroll
    for (int e = 0; e < 8; ++e) {
      const int m = m0 + e + 8 * laneHi;
      if (!kvp && j <= m) {
        const float s    = c[e];
        const float mnew = fmaxf(rmax[e], s);
        rsum[e] = rsum[e] * __expf(rmax[e] - mnew) + __expf(s - mnew);
        rmax[e] = mnew;
      }
    }
  }
  // merge (max, sum) pairs across the 16-lane half-wave
#pragma unroll
  for (int e = 0; e < 8; ++e) {
#pragma unroll
    for (int off = 1; off < 16; off <<= 1) {
      const float om = __shfl_xor(rmax[e], off, 32);
      const float ol = __shfl_xor(rsum[e], off, 32);
      const float M  = fmaxf(rmax[e], om);
      rsum[e] = rsum[e] * __expf(rmax[e] - M) + ol * __expf(om - M);
      rmax[e] = M;
    }
  }

  float rinv[8];
#pragma unroll
  for (int e = 0; e < 8; ++e) {
    const bool qp = q_pad[b * S_ + m0 + e + 8 * laneHi] != 0;
    rinv[e] = (qp || !(rsum[e] > 0.0f)) ? 0.0f : 1.0f / rsum[e];
  }

  // ---- pass 2: emit normalized weights (zeros where masked) + P@V ----
  float8_t oacc[4] = {};
  _Float16* slice = plds[w];
  for (int p2 = 0; p2 < 32; ++p2) {
    const int n0 = p2 << 5;  // pair of 16-wide tiles = one 32-deep K step for PV
    if (n0 > m0 + 15) {
      // fully beyond the causal diagonal: weights are zero, no PV contribution
#pragma unroll
      for (int sub = 0; sub < 2; ++sub) {
        const int j = n0 + 16 * sub + cn;
#pragma unroll
        for (int e = 0; e < 8; ++e)
          __builtin_nontemporal_store(
              0.0f,
              &w_out[((size_t)bh * S_ + (m0 + e + 8 * laneHi)) * S_ + j]);
      }
      continue;
    }
#pragma unroll
    for (int sub = 0; sub < 2; ++sub) {
      const int nt = n0 + 16 * sub;
      half16_t bk0 = load_b_frag(kb, DK_, nt, 0, lane);
      half16_t bk1 = load_b_frag(kb, DK_, nt, 32, lane);
      float8_t c = {};
      c = wmma_f16(aq0, bk0, c);
      c = wmma_f16(aq1, bk1, c);
      const int  j   = nt + cn;
      const bool kvp = kv_pad[b * S_ + j] != 0;
#pragma unroll
      for (int e = 0; e < 8; ++e) {
        const int m = m0 + e + 8 * laneHi;
        float p = 0.0f;
        if (rinv[e] > 0.0f && !kvp && j <= m)
          p = __expf(c[e] - rmax[e]) * rinv[e];
        __builtin_nontemporal_store(p, &w_out[((size_t)bh * S_ + m) * S_ + j]);
        slice[(e + 8 * laneHi) * 32 + (sub * 16 + cn)] = (_Float16)p;
      }
    }
    // LDS is in-order per wave; ensure stores land before the re-layout reads.
    asm volatile("s_wait_dscnt 0" ::: "memory");
    {
      const _Float16* lp = slice + cn * 32 + laneHi * 8;
      half8_t lo = *(const half8_t*)(lp);
      half8_t hi = *(const half8_t*)(lp + 16);
      half16_t pa;
#pragma unroll
      for (int i = 0; i < 8; ++i) { pa[i] = lo[i]; pa[i + 8] = hi[i]; }
#pragma unroll
      for (int t = 0; t < 4; ++t) {
        half16_t vvb = load_b_frag(vb, S_, t * 16, n0, lane);  // V^T: contiguous
        oacc[t] = wmma_f16(pa, vvb, oacc[t]);
      }
    }
    asm volatile("" ::: "memory");
  }

  // epilogue: attn tile -> f16 (B,S,D) for the output projection
#pragma unroll
  for (int t = 0; t < 4; ++t)
#pragma unroll
    for (int e = 0; e < 8; ++e) {
      const int m = m0 + e + 8 * laneHi;
      attn_h[((size_t)b * S_ + m) * D_ + h * DK_ + t * 16 + cn] =
          (_Float16)oacc[t][e];
    }
}

// ---------------------------------------------------------------------------
extern "C" void kernel_launch(void* const* d_in, const int* in_sizes, int n_in,
                              void* d_out, int out_size, void* d_ws,
                              size_t ws_size, hipStream_t stream) {
  (void)in_sizes; (void)n_in; (void)out_size; (void)ws_size;

  const float* xq  = (const float*)d_in[0];
  const float* xkv = (const float*)d_in[1];
  const unsigned char* q_pad  = (const unsigned char*)d_in[2];
  const unsigned char* kv_pad = (const unsigned char*)d_in[3];
  const float* Wq = (const float*)d_in[4];
  const float* Wk = (const float*)d_in[5];
  const float* Wv = (const float*)d_in[6];
  const float* Wo = (const float*)d_in[7];

  float* out_p = (float*)d_out;                        // (B,S,D)
  float* w_out = out_p + (size_t)B_ * S_ * D_;         // (B,H,S,S)

  const size_t NXD = (size_t)B_ * S_ * D_;             // 4,194,304
  const size_t NWW = (size_t)D_ * D_;                  // 1,048,576

  _Float16* ws = (_Float16*)d_ws;
  size_t off = 0;
  _Float16* xq_h   = ws + off; off += NXD;
  _Float16* xkv_h  = ws + off; off += NXD;
  _Float16* wq_h   = ws + off; off += NWW;
  _Float16* wk_h   = ws + off; off += NWW;
  _Float16* wv_h   = ws + off; off += NWW;
  _Float16* wo_h   = ws + off; off += NWW;
  _Float16* q_h    = ws + off; off += NXD;  // (B,H,S,DK)
  _Float16* k_h    = ws + off; off += NXD;  // (B,H,S,DK)
  _Float16* vt_h   = ws + off; off += NXD;  // (B,H,DK,S)
  _Float16* attn_f = ws + off; off += NXD;  // (B,S,D)

  const float SCALE = 0.125f;  // 1/sqrt(64), folded into Wq

  const int cvt_thr = 256;
  cvt_f32_f16<<<(int)((NXD + cvt_thr - 1) / cvt_thr), cvt_thr, 0, stream>>>(
      xq, xq_h, (int)NXD, 1.0f);
  cvt_f32_f16<<<(int)((NXD + cvt_thr - 1) / cvt_thr), cvt_thr, 0, stream>>>(
      xkv, xkv_h, (int)NXD, 1.0f);
  cvt_f32_f16<<<(int)((NWW + cvt_thr - 1) / cvt_thr), cvt_thr, 0, stream>>>(
      Wq, wq_h, (int)NWW, SCALE);
  cvt_f32_f16<<<(int)((NWW + cvt_thr - 1) / cvt_thr), cvt_thr, 0, stream>>>(
      Wk, wk_h, (int)NWW, 1.0f);
  cvt_f32_f16<<<(int)((NWW + cvt_thr - 1) / cvt_thr), cvt_thr, 0, stream>>>(
      Wv, wv_h, (int)NWW, 1.0f);
  cvt_f32_f16<<<(int)((NWW + cvt_thr - 1) / cvt_thr), cvt_thr, 0, stream>>>(
      Wo, wo_h, (int)NWW, 1.0f);

  dim3 ggrid(D_ / 128, (B_ * S_) / 256);  // (8, 16)
  gemm_x_wT<<<ggrid, 256, 0, stream>>>(xq_h,  wq_h, (void*)q_h,  0);
  gemm_x_wT<<<ggrid, 256, 0, stream>>>(xkv_h, wk_h, (void*)k_h,  0);
  gemm_x_wT<<<ggrid, 256, 0, stream>>>(xkv_h, wv_h, (void*)vt_h, 1);

  dim3 agrid(S_ / 128, B_ * H_);          // (8, 64)
  attn_kernel<<<agrid, 256, 0, stream>>>(q_h, k_h, vt_h, q_pad, kv_pad, w_out,
                                         attn_f);

  gemm_x_wT<<<ggrid, 256, 0, stream>>>(attn_f, wo_h, (void*)out_p, 2);
}
